// HybridWindowAttention_54348516164311
// MI455X (gfx1250) — matrix operand order repaired and checked by reading, not compiled, not run on hardware
//
#include <hip/hip_runtime.h>
#include <hip/hip_bf16.h>

// ---------------------------------------------------------------------------
// Fused Swin window attention for gfx1250 (MI455X).
//   grid = 2048 windows, block = 512 threads (16 wave32 waves, wave == head).
//   All intermediates (q,k,vT,probs,attn_out) stay in LDS / registers;
//   x is read from HBM exactly once (fragments hoisted into registers),
//   weights + combined bias/mask tables live in L2.
//   Matrix math: v_wmma_f32_16x16x32_f16 (fp32 accumulate).
// ---------------------------------------------------------------------------

typedef _Float16 h16 __attribute__((ext_vector_type(16)));
typedef _Float16 h8  __attribute__((ext_vector_type(8)));
typedef float    f8  __attribute__((ext_vector_type(8)));
typedef float    f4  __attribute__((ext_vector_type(4)));

#define NTOK  49
#define DIMC  512
#define NHEAD 16
#define NWIN  64

static __device__ inline h16 h16zero() {
  h16 r;
  #pragma unroll
  for (int i = 0; i < 16; ++i) r[i] = (_Float16)0.0f;
  return r;
}
static __device__ inline f8 f8zero() {
  f8 r;
  #pragma unroll
  for (int i = 0; i < 8; ++i) r[i] = 0.0f;
  return r;
}
static __device__ inline h16 h16cat(h8 lo, h8 hi) {
  return __builtin_shufflevector(lo, hi, 0,1,2,3,4,5,6,7,8,9,10,11,12,13,14,15);
}
// 16 contiguous halves (B-fragment: K split contiguously across lane halves)
static __device__ inline h16 ldh16(const _Float16* p) {
  const h8* q = (const h8*)p;
  return h16cat(q[0], q[1]);
}
// A-fragment pattern: halves [0..15] relative to p
static __device__ inline h16 ldh8x2(const _Float16* p) {
  const h8* q = (const h8*)p;
  return h16cat(q[0], q[1]);
}
// A-fragment from fp32 row: floats [0..15] relative to row+koff
static __device__ inline h16 ldx_frag(const float* row, int koff) {
  const f4* p = (const f4*)(row + koff);
  f4 x0 = p[0], x1 = p[1], x2 = p[2], x3 = p[3];
  h16 a;
  #pragma unroll
  for (int i = 0; i < 4; ++i) {
    a[i]      = (_Float16)x0[i];
    a[4 + i]  = (_Float16)x1[i];
    a[8 + i]  = (_Float16)x2[i];
    a[12 + i] = (_Float16)x3[i];
  }
  return a;
}
static __device__ inline f8 wmma32(h16 a, h16 b, f8 c) {
  return __builtin_amdgcn_wmma_f32_16x16x32_f16(false, a, false, b, (short)0, c,
                                                false, false);
}

// ---------------------------------------------------------------------------
// Prolog: f16-transpose weights + precompute combined bias/mask table
//   WqkvT  [1536][512] f16   (qkv_w^T)
//   WprojT [ 512][512] f16   (proj_w^T)
//   BM     [64][16][49][49] fp32 :  bias_table[rel(m,n)][h] + mask[wi][m][n]
// ---------------------------------------------------------------------------
__global__ void swin_prolog(const float* __restrict__ qkv_w,
                            const float* __restrict__ proj_w,
                            const float* __restrict__ bias_table,
                            const float* __restrict__ mask,
                            _Float16* __restrict__ WqkvT,
                            _Float16* __restrict__ WprojT,
                            float* __restrict__ BM) {
  const size_t stride = (size_t)gridDim.x * blockDim.x;
  const size_t i0 = (size_t)blockIdx.x * blockDim.x + threadIdx.x;
  for (size_t idx = i0; idx < (size_t)1536 * 512; idx += stride) {
    size_t n = idx >> 9, k = idx & 511;
    WqkvT[idx] = (_Float16)qkv_w[k * 1536 + n];
  }
  for (size_t idx = i0; idx < (size_t)512 * 512; idx += stride) {
    size_t n = idx >> 9, k = idx & 511;
    WprojT[idx] = (_Float16)proj_w[k * 512 + n];
  }
  for (size_t idx = i0; idx < (size_t)NWIN * NHEAD * NTOK * NTOK; idx += stride) {
    const int n  = (int)(idx % NTOK);
    size_t t     = idx / NTOK;
    const int m  = (int)(t % NTOK);
    t /= NTOK;
    const int h  = (int)(t & (NHEAD - 1));
    const int wi = (int)(t >> 4);
    const int my = m / 7, mxx = m - my * 7;
    const int ny = n / 7, nxx = n - ny * 7;
    const int ri = (my - ny + 6) * 13 + (mxx - nxx + 6);
    BM[idx] = bias_table[ri * NHEAD + h] +
              mask[((size_t)wi * NTOK + m) * NTOK + n];
  }
}

// ---------------------------------------------------------------------------
// Fused attention kernel
// ---------------------------------------------------------------------------
__global__ __launch_bounds__(512, 1)
void swin_attn_kernel(const float* __restrict__ x,
                      const float* __restrict__ qkv_b,
                      const float* __restrict__ proj_b,
                      const float* __restrict__ BM,
                      const _Float16* __restrict__ WqkvT,
                      const _Float16* __restrict__ WprojT,
                      float* __restrict__ out) {
  // per-wave slab: q[49][32] | k[49][32] (reused as P scratch [16][64]) | vT[32][52]
  __shared__ __align__(16) _Float16 pool[16 * 4800];  // 153,600 B

  const int tid = threadIdx.x;
  const int w   = tid >> 5;          // wave id == head id
  const int l   = tid & 31;
  const int l16 = l & 15;
  const int hh  = l >> 4;            // lane half
  const int blk = blockIdx.x;
  const int wi  = blk & (NWIN - 1);  // window-mask index (b % nW)
  const float scale = 0.17677669529663687f;  // 1/sqrt(32)

  _Float16* slab = pool + w * 4800;
  _Float16* qs   = slab;             // [49][32], q pre-scaled
  _Float16* ks_  = slab + 1568;      // [49][32]
  _Float16* vT   = slab + 3136;      // [32][52] (transposed v)

  const float* xwin = x + (size_t)blk * NTOK * DIMC;
  const float* BMh  = BM + (size_t)(wi * NHEAD + w) * NTOK * NTOK;

  // ---------------- QKV projection (each wave: its own head's q,k,v) -------
  // x A-fragments hoisted per row-tile: one HBM pass over x; the 6 weight
  // column groups (q/k/v x 2 subtiles) all reuse them from registers.
  for (int mt = 0; mt < 4; ++mt) {
    const int m_a = mt * 16 + l16;
    const bool mv = (m_a < NTOK);
    const float* xrow = xwin + (size_t)(mv ? m_a : 0) * DIMC + (hh ? 16 : 0);
    h16 xa[16];
    #pragma unroll
    for (int ksI = 0; ksI < 16; ++ksI)
      xa[ksI] = mv ? ldx_frag(xrow, ksI * 32) : h16zero();

    for (int s = 0; s < 3; ++s) {           // 0=q 1=k 2=v
      for (int sub = 0; sub < 2; ++sub) {   // two 16-col subtiles of hd=32
        f8 acc = f8zero();
        const int ncol = s * 512 + w * 32 + sub * 16;
        const _Float16* wrow = WqkvT + (size_t)(ncol + l16) * 512 + (hh ? 16 : 0);
        #pragma unroll
        for (int ksI = 0; ksI < 16; ++ksI)
          acc = wmma32(xa[ksI], ldh16(wrow + ksI * 32), acc);
        const float bval = qkv_b[ncol + l16];
        const int f = sub * 16 + l16;
        #pragma unroll
        for (int r = 0; r < 8; ++r) {
          const int m = mt * 16 + r + hh * 8;
          if (m < NTOK) {
            const float v = acc[r] + bval;
            if (s == 0)      qs[m * 32 + f]  = (_Float16)(v * scale);
            else if (s == 1) ks_[m * 32 + f] = (_Float16)v;
            else             vT[f * 52 + m]  = (_Float16)v;
          }
        }
      }
    }
  }
  asm volatile("s_wait_dscnt 0" ::: "memory");  // same-wave LDS RAW

  // ---------------- load K^T / V fragments into registers ------------------
  h16 kf[4];
  #pragma unroll
  for (int nt = 0; nt < 4; ++nt) {
    const int tok = nt * 16 + l16;
    kf[nt] = (tok < NTOK) ? ldh16(ks_ + tok * 32 + (hh ? 16 : 0)) : h16zero();
  }
  h16 vf[2][2];
  #pragma unroll
  for (int k2 = 0; k2 < 2; ++k2) {
    #pragma unroll
    for (int vt = 0; vt < 2; ++vt) {
      const int f = vt * 16 + l16;
      h16 b = h16zero();
      #pragma unroll
      for (int i = 0; i < 16; ++i) {
        const int t = k2 * 32 + (hh ? 16 : 0) + i;
        if (t < NTOK) b[i] = vT[f * 52 + t];
      }
      vf[k2][vt] = b;
    }
  }

  // ---------------- logits -> softmax -> prob A-fragments ------------------
  h16 pa[4][2];
  for (int mt = 0; mt < 4; ++mt) {
    const int mq = mt * 16 + l16;
    h16 qa = (mq < NTOK) ? ldh8x2(qs + mq * 32 + (hh ? 16 : 0)) : h16zero();
    f8 acc[4];
    #pragma unroll
    for (int nt = 0; nt < 4; ++nt) acc[nt] = wmma32(qa, kf[nt], f8zero());

    // add precomputed (relative-position bias + window mask)
    #pragma unroll
    for (int r = 0; r < 8; ++r) {
      const int m = mt * 16 + r + hh * 8;        // query token
      const bool mvalid = m < NTOK;
      const float* bmrow = BMh + m * NTOK;
      #pragma unroll
      for (int nt = 0; nt < 4; ++nt) {
        const int n = nt * 16 + l16;             // key token
        float v;
        if (!mvalid)        v = 0.0f;
        else if (n >= NTOK) v = -1e30f;
        else                v = acc[nt][r] + bmrow[n];
        acc[nt][r] = v;
      }
    }
    // softmax per row (row lives in the 16 lanes of one half-wave)
    #pragma unroll
    for (int r = 0; r < 8; ++r) {
      float mx = fmaxf(fmaxf(acc[0][r], acc[1][r]), fmaxf(acc[2][r], acc[3][r]));
      #pragma unroll
      for (int off = 1; off < 16; off <<= 1) mx = fmaxf(mx, __shfl_xor(mx, off, 32));
      float sum = 0.0f;
      #pragma unroll
      for (int nt = 0; nt < 4; ++nt) {
        const float p = __expf(acc[nt][r] - mx);
        acc[nt][r] = p;
        sum += p;
      }
      #pragma unroll
      for (int off = 1; off < 16; off <<= 1) sum += __shfl_xor(sum, off, 32);
      const float inv = __builtin_amdgcn_rcpf(sum);
      #pragma unroll
      for (int nt = 0; nt < 4; ++nt) acc[nt][r] *= inv;
    }
    // C-layout -> A-layout via per-wave scratch (reuses dead k region)
    _Float16* P = ks_;  // [16][64]
    #pragma unroll
    for (int r = 0; r < 8; ++r) {
      const int ml = r + hh * 8;
      #pragma unroll
      for (int nt = 0; nt < 4; ++nt)
        P[ml * 64 + nt * 16 + l16] = (_Float16)acc[nt][r];
    }
    asm volatile("s_wait_dscnt 0" ::: "memory");
    #pragma unroll
    for (int k2 = 0; k2 < 2; ++k2)
      pa[mt][k2] = ldh8x2(P + l16 * 64 + k2 * 32 + (hh ? 16 : 0));
  }

  __syncthreads();  // all slab reads done; attn_out overlays the slabs now

  // ---------------- out = P @ V  -> attn_out LDS [49][512] f16 -------------
  _Float16* AO = pool;
  #pragma unroll
  for (int mt = 0; mt < 4; ++mt) {
    #pragma unroll
    for (int vt = 0; vt < 2; ++vt) {
      f8 acc = f8zero();
      #pragma unroll
      for (int k2 = 0; k2 < 2; ++k2) acc = wmma32(pa[mt][k2], vf[k2][vt], acc);
      #pragma unroll
      for (int r = 0; r < 8; ++r) {
        const int m = mt * 16 + r + hh * 8;
        if (m < NTOK)
          AO[m * 512 + w * 32 + vt * 16 + l16] = (_Float16)acc[r];
      }
    }
  }

  __syncthreads();

  // ---------------- output projection --------------------------------------
  for (int t = w; t < 128; t += 16) {  // 4 row tiles x 32 col tiles
    const int nt = t >> 2, mt = t & 3;
    f8 acc = f8zero();
    const int ma = mt * 16 + l16;
    const bool mv = ma < NTOK;
    const _Float16* arow = AO + (mv ? ma : 0) * 512 + (hh ? 16 : 0);
    const _Float16* wrow = WprojT + (size_t)(nt * 16 + l16) * 512 + (hh ? 16 : 0);
    #pragma unroll
    for (int ksI = 0; ksI < 16; ++ksI) {
      h16 a = mv ? ldh8x2(arow + ksI * 32) : h16zero();
      h16 b = ldh16(wrow + ksI * 32);
      acc = wmma32(a, b, acc);
    }
    const float pbv = proj_b[nt * 16 + l16];
    #pragma unroll
    for (int r = 0; r < 8; ++r) {
      const int m = mt * 16 + r + hh * 8;
      if (m < NTOK)
        out[((size_t)blk * NTOK + m) * DIMC + nt * 16 + l16] = acc[r] + pbv;
    }
  }
}

// ---------------------------------------------------------------------------
extern "C" void kernel_launch(void* const* d_in, const int* in_sizes, int n_in,
                              void* d_out, int out_size, void* d_ws, size_t ws_size,
                              hipStream_t stream) {
  const float* x          = (const float*)d_in[0];
  const float* mask       = (const float*)d_in[1];
  const float* qkv_w      = (const float*)d_in[2];
  const float* qkv_b      = (const float*)d_in[3];
  const float* proj_w     = (const float*)d_in[4];
  const float* proj_b     = (const float*)d_in[5];
  const float* bias_table = (const float*)d_in[6];

  _Float16* WqkvT  = (_Float16*)d_ws;                  // [1536][512] f16
  _Float16* WprojT = WqkvT + (size_t)1536 * 512;       // [512][512]  f16
  float*    BM     = (float*)((char*)d_ws + 2 * 1024 * 1024);  // [64][16][49][49] fp32

  swin_prolog<<<512, 256, 0, stream>>>(qkv_w, proj_w, bias_table, mask,
                                       WqkvT, WprojT, BM);
  swin_attn_kernel<<<2048, 512, 0, stream>>>(x, qkv_b, proj_b, BM,
                                             WqkvT, WprojT, (float*)d_out);
}
